// SpasreMultiscaleEncoderTest_13993003450855
// MI455X (gfx1250) — matrix operand order repaired and checked by reading, not compiled, not run on hardware
//
#include <hip/hip_runtime.h>

#define ZDIM 21
#define YDIM 256
#define XDIM 256
#define BDIM 2
#define CINC 16
#define C1C  32
#define C2C  64
#define GRID_CELLS (BDIM*ZDIM*YDIM*XDIM)
#define BN_EPS 1e-3f

typedef __attribute__((ext_vector_type(16))) __bf16 v16bf;
typedef __attribute__((ext_vector_type(8)))  float v8f;

union Frag {
    v16bf v;
    unsigned u[8];
    uint4 q[2];
};

__device__ __forceinline__ unsigned short f2bf(float f) {
    unsigned u = __float_as_uint(f);
    u += 0x7FFFu + ((u >> 16) & 1u);     // round-to-nearest-even
    return (unsigned short)(u >> 16);
}
__device__ __forceinline__ unsigned packbf(float a, float b) {
    return (unsigned)f2bf(a) | ((unsigned)f2bf(b) << 16);
}

// ---------------- rulebook hash ----------------
__global__ void init_grid_k(int* __restrict__ grid) {
    int i = blockIdx.x * blockDim.x + threadIdx.x;
    if (i < GRID_CELLS) grid[i] = -1;
}
__global__ void scatter_k(const int* __restrict__ coors, int* __restrict__ grid, int N) {
    int i = blockIdx.x * blockDim.x + threadIdx.x;
    if (i >= N) return;
    int b = coors[i*4+0], z = coors[i*4+1], y = coors[i*4+2], x = coors[i*4+3];
    grid[((b*ZDIM + z)*YDIM + y)*XDIM + x] = i;
}

// ---------------- one-time f32 -> bf16 conversion of voxel features ----------------
__global__ void prep_feat_k(const float* __restrict__ feat, unsigned* __restrict__ featbf,
                            int total_u32) {
    int t = blockIdx.x * blockDim.x + threadIdx.x;       // one u32 = 2 channels
    if (t >= total_u32) return;
    featbf[t] = packbf(feat[2*t], feat[2*t + 1]);
}

// ---------------- fold BN into per-channel scale/shift ----------------
__global__ void prep_bn_k(const float* __restrict__ g, const float* __restrict__ b,
                          const float* __restrict__ m, const float* __restrict__ v,
                          float* __restrict__ ss, int C) {
    int i = blockIdx.x * blockDim.x + threadIdx.x;
    if (i >= C) return;
    float s = g[i] * rsqrtf(v[i] + BN_EPS);
    ss[i]     = s;              // scale
    ss[i + C] = b[i] - m[i]*s;  // shift
}

// ---------------- weight swizzle into WMMA B-fragment layout ----------------
// B layout (16-bit, KxN=32x16): VGPR v, lane L: n = L&15, s = 2v+h, k = (s&8)*2 + (L<16?0:8) + (s&7)
__global__ void prep_w1_k(const float* __restrict__ W1, unsigned* __restrict__ w1w) {
    int t = blockIdx.x * blockDim.x + threadIdx.x;
    if (t >= 14*2*32*8) return;
    int v = t & 7, lane = (t >> 3) & 31, ntile = (t >> 8) & 1, pair = t >> 9;
    int n = ntile*16 + (lane & 15);
    float vals[2];
    for (int h = 0; h < 2; ++h) {
        int k = ((2*v + h) & 8)*2 + ((lane & 16) ? 8 : 0) + ((2*v + h) & 7);
        int p = k >> 4, c = k & 15;
        int off = pair*2 + p;
        vals[h] = (off < 27) ? W1[(off*CINC + c)*C1C + n] : 0.f;
    }
    w1w[t] = packbf(vals[0], vals[1]);
}
__global__ void prep_w2_k(const float* __restrict__ W2, unsigned* __restrict__ w2w) {
    int t = blockIdx.x * blockDim.x + threadIdx.x;
    if (t >= 27*4*32*8) return;
    int v = t & 7, lane = (t >> 3) & 31, ntile = (t >> 8) & 3, off = t >> 10;
    int n = ntile*16 + (lane & 15);
    int k0 = ((2*v) & 8)*2 + ((lane & 16) ? 8 : 0) + ((2*v) & 7);
    float a = W2[(off*C1C + k0    )*C2C + n];
    float b = W2[(off*C1C + k0 + 1)*C2C + n];
    w2w[t] = packbf(a, b);
}

// ---------------- stage 1: SubMConv3d 16->32 + BN + ReLU, bf16 out ----------------
__global__ __launch_bounds__(256) void stage1_k(
    const unsigned short* __restrict__ featbf, const int* __restrict__ coors,
    const int* __restrict__ grid, const unsigned* __restrict__ w1w,
    const float* __restrict__ bn1,
    unsigned short* __restrict__ f1bf, int N)
{
    int wave = threadIdx.x >> 5;
    int lane = threadIdx.x & 31;
    int row0 = (blockIdx.x * 8 + wave) * 16;
    if (row0 >= N) return;                    // wave-uniform exit
    int m  = lane & 15;
    int hh = lane >> 4;                       // half-wave: channel half owned
    int row = row0 + m;
    bool vrow = row < N;
    int rl = vrow ? row : 0;
    int cb = coors[rl*4+0], cz = coors[rl*4+1], cy = coors[rl*4+2], cx = coors[rl*4+3];

    v8f acc0 = {}; v8f acc1 = {};
    for (int pair = 0; pair < 14; ++pair) {
        // ---- issue both B-fragment load groups first (batched, partial waits) ----
        const uint4* bp0 = (const uint4*)(w1w + ((pair*2 + 0)*32 + lane)*8);
        const uint4* bp1 = (const uint4*)(w1w + ((pair*2 + 1)*32 + lane)*8);
        Frag fb0, fb1;
        fb0.q[0] = bp0[0]; fb0.q[1] = bp0[1];
        fb1.q[0] = bp1[0]; fb1.q[1] = bp1[1];

        // ---- gather A fragment (two offsets packed along K) ----
        Frag a;
        #pragma unroll
        for (int p = 0; p < 2; ++p) {
            int off = pair*2 + p;
            int idx = -1;
            if (off < 27 && vrow) {
                int dz = off/9 - 1, dy = (off/3)%3 - 1, dx = off%3 - 1;
                int z = cz + dz, y = cy + dy, x = cx + dx;
                if ((unsigned)z < ZDIM && (unsigned)y < YDIM && (unsigned)x < XDIM)
                    idx = grid[((cb*ZDIM + z)*YDIM + y)*XDIM + x];
            }
            if (idx >= 0)
                a.q[p] = *(const uint4*)(featbf + (size_t)idx*CINC + hh*8);
            else
                a.q[p] = make_uint4(0,0,0,0);
        }

        acc0 = __builtin_amdgcn_wmma_f32_16x16x32_bf16(false, a.v, false, fb0.v,
                                                       (short)0, acc0, false, false);
        acc1 = __builtin_amdgcn_wmma_f32_16x16x32_bf16(false, a.v, false, fb1.v,
                                                       (short)0, acc1, false, false);
    }
    #pragma unroll
    for (int t = 0; t < 2; ++t) {
        int n = t*16 + m;
        float s  = bn1[n];
        float sh = bn1[n + C1C];
        v8f acc = (t == 0) ? acc0 : acc1;
        #pragma unroll
        for (int v = 0; v < 8; ++v) {
            int orow = row0 + v + hh*8;
            if (orow < N) {
                float val = acc[v]*s + sh;
                val = val > 0.f ? val : 0.f;
                f1bf[(size_t)orow*C1C + n] = f2bf(val);
            }
        }
    }
}

// ---------------- stage 2: strided SparseConv3d 32->64 + BN + ReLU ----------------
__global__ __launch_bounds__(256) void stage2_k(
    const unsigned short* __restrict__ f1bf, const int* __restrict__ ocoors,
    const int* __restrict__ grid, const unsigned* __restrict__ w2w,
    const float* __restrict__ bn2,
    float* __restrict__ out, int M)
{
    int wave = threadIdx.x >> 5;
    int lane = threadIdx.x & 31;
    int row0 = (blockIdx.x * 8 + wave) * 16;
    if (row0 >= M) return;
    int m  = lane & 15;
    int hh = lane >> 4;
    int row = row0 + m;
    bool vrow = row < M;
    int rl = vrow ? row : 0;
    int cb = ocoors[rl*4+0], oz = ocoors[rl*4+1], oy = ocoors[rl*4+2], ox = ocoors[rl*4+3];

    v8f acc[4];
    #pragma unroll
    for (int t = 0; t < 4; ++t) acc[t] = (v8f){};

    for (int off = 0; off < 27; ++off) {
        // ---- issue all four B-fragment load groups first ----
        Frag fb[4];
        #pragma unroll
        for (int t = 0; t < 4; ++t) {
            const uint4* bp = (const uint4*)(w2w + ((off*4 + t)*32 + lane)*8);
            fb[t].q[0] = bp[0];
            fb[t].q[1] = bp[1];
        }

        // ---- gather A fragment ----
        int jz = off/9, jy = (off/3)%3, jx = off%3;
        int z = oz*2     + jz;      // pad 0 in z
        int y = oy*2 - 1 + jy;      // pad 1 in y
        int x = ox*2 - 1 + jx;      // pad 1 in x
        int idx = -1;
        if (vrow && (unsigned)z < ZDIM && (unsigned)y < YDIM && (unsigned)x < XDIM)
            idx = grid[((cb*ZDIM + z)*YDIM + y)*XDIM + x];
        Frag a;
        if (idx >= 0) {
            a.q[0] = *(const uint4*)(f1bf + (size_t)idx*C1C      + hh*8);
            a.q[1] = *(const uint4*)(f1bf + (size_t)idx*C1C + 16 + hh*8);
        } else {
            a.q[0] = make_uint4(0,0,0,0);
            a.q[1] = make_uint4(0,0,0,0);
        }

        #pragma unroll
        for (int t = 0; t < 4; ++t) {
            acc[t] = __builtin_amdgcn_wmma_f32_16x16x32_bf16(false, a.v, false, fb[t].v,
                                                             (short)0, acc[t], false, false);
        }
    }
    #pragma unroll
    for (int t = 0; t < 4; ++t) {
        int n = t*16 + m;
        float s  = bn2[n];
        float sh = bn2[n + C2C];
        #pragma unroll
        for (int v = 0; v < 8; ++v) {
            int orow = row0 + v + hh*8;
            if (orow < M) {
                float val = acc[t][v]*s + sh;
                out[(size_t)orow*C2C + n] = val > 0.f ? val : 0.f;
            }
        }
    }
}

// ---------------- tail: out_coors + batch_size appended after f2 ----------------
__global__ void tail_k(const int* __restrict__ ocoors, const int* __restrict__ bsz,
                       int* __restrict__ outtail, int M) {
    int i = blockIdx.x * blockDim.x + threadIdx.x;
    if (i < M*4)       outtail[i] = ocoors[i];
    else if (i == M*4) outtail[i] = bsz[0];
}

extern "C" void kernel_launch(void* const* d_in, const int* in_sizes, int n_in,
                              void* d_out, int out_size, void* d_ws, size_t ws_size,
                              hipStream_t stream) {
    const float* feat   = (const float*)d_in[0];
    const int*   coors  = (const int*)  d_in[1];
    const int*   ocoors = (const int*)  d_in[2];
    const float* W1 = (const float*)d_in[3];
    const float* g1 = (const float*)d_in[4];
    const float* b1 = (const float*)d_in[5];
    const float* m1 = (const float*)d_in[6];
    const float* v1 = (const float*)d_in[7];
    const float* W2 = (const float*)d_in[8];
    const float* g2 = (const float*)d_in[9];
    const float* b2 = (const float*)d_in[10];
    const float* m2 = (const float*)d_in[11];
    const float* v2 = (const float*)d_in[12];
    const int*   bsz = (const int*) d_in[13];

    int N = in_sizes[0] / CINC;
    int M = in_sizes[2] / 4;

    // workspace layout
    char* ws = (char*)d_ws;
    int*            grid   = (int*)ws;                                 // 2,752,512 i32
    unsigned*       w1w    = (unsigned*)(ws + (size_t)GRID_CELLS*4);   // 7,168 u32
    unsigned*       w2w    = w1w + 14*2*32*8;                          // 27,648 u32
    float*          bn1    = (float*)(w2w + 27*4*32*8);                // 64 f32
    float*          bn2    = bn1 + 2*C1C;                              // 128 f32
    unsigned*       featbf = (unsigned*)(bn2 + 2*C2C);                 // N*8 u32 (bf16)
    unsigned short* f1bf   = (unsigned short*)(featbf + (size_t)N*8);  // N*32 bf16

    init_grid_k<<<(GRID_CELLS + 255)/256, 256, 0, stream>>>(grid);
    scatter_k  <<<(N + 255)/256,          256, 0, stream>>>(coors, grid, N);
    int fu32 = N * 8;
    prep_feat_k<<<(fu32 + 255)/256,       256, 0, stream>>>(feat, featbf, fu32);
    prep_w1_k  <<<(14*2*32*8 + 255)/256,  256, 0, stream>>>(W1, w1w);
    prep_w2_k  <<<(27*4*32*8 + 255)/256,  256, 0, stream>>>(W2, w2w);
    prep_bn_k  <<<1, C1C, 0, stream>>>(g1, b1, m1, v1, bn1, C1C);
    prep_bn_k  <<<1, C2C, 0, stream>>>(g2, b2, m2, v2, bn2, C2C);

    int tiles1 = (N + 15) / 16;
    stage1_k<<<(tiles1 + 7)/8, 256, 0, stream>>>((const unsigned short*)featbf,
                                                 coors, grid, w1w, bn1, f1bf, N);
    int tiles2 = (M + 15) / 16;
    stage2_k<<<(tiles2 + 7)/8, 256, 0, stream>>>(f1bf, ocoors, grid, w2w,
                                                 bn2, (float*)d_out, M);
    tail_k<<<(M*4 + 1 + 255)/256, 256, 0, stream>>>(ocoors, bsz,
                                                    (int*)d_out + (size_t)M*C2C, M);
}
